// Citeseer_Method_50697793961993
// MI455X (gfx1250) — matrix operand order repaired and checked by reading, not compiled, not run on hardware
//
#include <hip/hip_runtime.h>
#include <hip/hip_bf16.h>

#define N_NODES 50000
#define N_EDGES 400000
#define K_IN    3703
#define K_PAD   3712      // 116 * 32
#define NT      (K_PAD / 32)
#define DIM     256
#define NOUT    6

typedef __attribute__((ext_vector_type(16))) __bf16 v16bf;
typedef __attribute__((ext_vector_type(2)))  __bf16 v2bf;
typedef __attribute__((ext_vector_type(8)))  float  v8f;
typedef __attribute__((ext_vector_type(4)))  int    v4i;

#if __has_builtin(__builtin_amdgcn_global_load_async_to_lds_b128)
#define USE_ASYNC_B 1
typedef __attribute__((address_space(1))) v4i gv4i_t;   // global AS pointer elem
typedef __attribute__((address_space(3))) v4i lv4i_t;   // LDS AS pointer elem
#else
#define USE_ASYNC_B 0
#endif

__device__ __forceinline__ void wait_async0() {
#if USE_ASYNC_B
#if __has_builtin(__builtin_amdgcn_s_wait_asynccnt)
    __builtin_amdgcn_s_wait_asynccnt(0);
#else
    asm volatile("s_wait_asynccnt 0" ::: "memory");
#endif
#endif
}

__device__ __forceinline__ unsigned short bfbits(float f) {
    union { float f; unsigned u; } in; in.f = f;
    unsigned r = in.u + 0x7FFFu + ((in.u >> 16) & 1u);   // round-nearest-even
    return (unsigned short)(r >> 16);
}

__device__ __forceinline__ __bf16 f2bf(float f) {
    union { unsigned short s; __bf16 b; } out;
    out.s = bfbits(f);
    return out.b;
}

// packed f32x2 -> bf16x2 as one uint (v_cvt_pk_bf16_f32 when available)
__device__ __forceinline__ unsigned pack2(float a, float b) {
#if __has_builtin(__builtin_amdgcn_cvt_pk_bf16_f32)
    union { v2bf v; unsigned u; } cv;
    cv.v = __builtin_amdgcn_cvt_pk_bf16_f32(a, b);
    return cv.u;
#else
    return (unsigned)bfbits(a) | ((unsigned)bfbits(b) << 16);
#endif
}

__device__ __forceinline__ v16bf mkfrag(uint4 lo, uint4 hi) {
    union { uint4 q[2]; v16bf v; } u;
    u.q[0] = lo; u.q[1] = hi;
    return u.v;
}

// -------- W1 fp32 [3703,256] -> bf16 transposed+padded [256][3712] ----------
__global__ void k_convert_w1(const float* __restrict__ W1, __bf16* __restrict__ w1bT) {
    int idx = blockIdx.x * 256 + threadIdx.x;          // over DIM*K_PAD
    if (idx >= DIM * K_PAD) return;
    int n = idx / K_PAD, k = idx % K_PAD;
    float v = (k < K_IN) ? W1[k * DIM + n] : 0.0f;
    w1bT[idx] = f2bf(v);
}

// ---------------- generic zero ----------------
__global__ void k_zero(float* __restrict__ p, int n) {
    int idx = blockIdx.x * 256 + threadIdx.x;
    if (idx < n) p[idx] = 0.0f;
}

// ---- GEMM1: support[N,256] = x[N,3703] @ W1  (bf16 WMMA, f32 accum) --------
// Block tile 128 x 256 (full N width -> x streamed from HBM exactly once).
// 256 threads = 8 waves arranged 2(M) x 4(N); wave tile 64x64 = acc[4][4].
// Double-buffered LDS; A converted f32->bf16 in flight (packed cvt);
// B async-copied to LDS from the pre-transposed bf16 weights.
__global__ void __launch_bounds__(256)
k_gemm1(const float* __restrict__ x, const __bf16* __restrict__ w1bT,
        float* __restrict__ support) {
    __shared__ __bf16 sA[2][128][32];   // [m][k]
    __shared__ __bf16 sB[2][256][32];   // [n][k] (K-contiguous per column)

    const int tid  = threadIdx.x;
    const int lane = tid & 31;
    const int wid  = tid >> 5;
    const int wm   = wid & 1;          // 2 M groups of 64 rows
    const int wn   = wid >> 1;         // 4 N groups of 64 cols
    const int ln15 = lane & 15;
    const int hi16 = lane >> 4;
    const int abase = hi16 ? 8 : 0;
    const int bbase = hi16 ? 8 : 0;

    const int m0 = blockIdx.x * 128;

    // ---- staging helpers (branch-free; no exec divergence) ----
    auto stageA_main = [&](int kt, int b) {
        const int k0 = kt * 32;
#pragma unroll
        for (int i = 0; i < 4; ++i) {
            int chunk = tid + i * 256;            // 0..1023
            int r  = chunk >> 3;
            int c4 = (chunk & 7) << 2;
            long m = (long)min(m0 + r, N_NODES - 1);
            const float* px = x + m * K_IN + k0 + c4;
            float v0 = px[0], v1 = px[1], v2 = px[2], v3 = px[3];
            uint2 pv; pv.x = pack2(v0, v1); pv.y = pack2(v2, v3);
            *(uint2*)(&sA[b][r][c4]) = pv;
        }
    };
    auto stageA_tail = [&](int b) {               // kt = NT-1, k0 = 3680
        const int k0 = (NT - 1) * 32;
#pragma unroll
        for (int i = 0; i < 4; ++i) {
            int chunk = tid + i * 256;
            int r  = chunk >> 3;
            int c4 = (chunk & 7) << 2;
            long m = (long)min(m0 + r, N_NODES - 1);
            const float* px = x + m * K_IN;
            float v[4];
#pragma unroll
            for (int j = 0; j < 4; ++j) {
                int k = k0 + c4 + j;
                float t = px[min(k, K_IN - 1)];   // clamped addr, uncond load
                v[j] = (k < K_IN) ? t : 0.0f;     // v_cndmask
            }
            uint2 pv; pv.x = pack2(v[0], v[1]); pv.y = pack2(v[2], v[3]);
            *(uint2*)(&sA[b][r][c4]) = pv;
        }
    };
    auto stageB = [&](int kt, int b) {
        const int k0 = kt * 32;
        // one thread copies one full n-row of the k-slice: 32 bf16 = 64 bytes
        const __bf16* gp = w1bT + (long)tid * K_PAD + k0;
        __bf16* lp = &sB[b][tid][0];
#if USE_ASYNC_B
        __builtin_amdgcn_global_load_async_to_lds_b128((gv4i_t*)gp, (lv4i_t*)lp, 0, 0);
        __builtin_amdgcn_global_load_async_to_lds_b128((gv4i_t*)gp, (lv4i_t*)lp, 16, 0);
        __builtin_amdgcn_global_load_async_to_lds_b128((gv4i_t*)gp, (lv4i_t*)lp, 32, 0);
        __builtin_amdgcn_global_load_async_to_lds_b128((gv4i_t*)gp, (lv4i_t*)lp, 48, 0);
#else
#pragma unroll
        for (int q = 0; q < 4; ++q)
            *((uint4*)lp + q) = *((const uint4*)gp + q);
#endif
    };

    v8f acc[4][4];
#pragma unroll
    for (int i = 0; i < 4; ++i)
#pragma unroll
        for (int j = 0; j < 4; ++j)
#pragma unroll
            for (int r = 0; r < 8; ++r) acc[i][j][r] = 0.0f;

    // prologue: stage tile 0
    stageA_main(0, 0);
    stageB(0, 0);

    for (int kt = 0; kt < NT; ++kt) {
        const int b = kt & 1;
        wait_async0();
        __syncthreads();                  // staged tile kt now visible

        if (kt + 1 < NT) {                // stage next tile into alternate buf
            if (kt + 1 == NT - 1) stageA_tail(b ^ 1);
            else                  stageA_main(kt + 1, b ^ 1);
            stageB(kt + 1, b ^ 1);
            if (kt + 2 < NT) {            // prefetch tile kt+2 of x (global_prefetch_b8)
                int m = min(m0 + (tid >> 1), N_NODES - 1);
                __builtin_prefetch(&x[(long)m * K_IN + (kt + 2) * 32], 0, 0);
            }
        }

        // fragments from current buffer
        v16bf af[4], bfr[4];
#pragma unroll
        for (int i = 0; i < 4; ++i) {
            const __bf16* row = &sA[b][wm * 64 + i * 16 + ln15][0];
            af[i] = mkfrag(*(const uint4*)(row + abase),
                           *(const uint4*)(row + abase + 16));
        }
#pragma unroll
        for (int j = 0; j < 4; ++j) {
            const __bf16* col = &sB[b][wn * 64 + j * 16 + ln15][0];
            bfr[j] = mkfrag(*(const uint4*)(col + bbase),
                            *(const uint4*)(col + bbase + 16));
        }
#pragma unroll
        for (int i = 0; i < 4; ++i)
#pragma unroll
            for (int j = 0; j < 4; ++j)
                acc[i][j] = __builtin_amdgcn_wmma_f32_16x16x32_bf16(
                    false, af[i], false, bfr[j], (short)0, acc[i][j],
                    false, false);
    }

    // epilogue: C/D layout -> row r + 8*hi16, col = lane&15 (coalesced b32s)
    const int rbase = m0 + wm * 64;
    const int cbase = wn * 64;
#pragma unroll
    for (int i = 0; i < 4; ++i)
#pragma unroll
        for (int j = 0; j < 4; ++j) {
            int col = cbase + j * 16 + ln15;
#pragma unroll
            for (int r = 0; r < 8; ++r) {
                int row = rbase + i * 16 + r + (hi16 ? 8 : 0);
                if (row < N_NODES) support[(long)row * DIM + col] = acc[i][j][r];
            }
        }
}

// ---------------- scatter layer 1: agg[dst] += support[src] * w ------------
__global__ void k_scatter1(const float* __restrict__ support,
                           const int* __restrict__ src, const int* __restrict__ dst,
                           const float* __restrict__ w, float* __restrict__ agg) {
    int gid = blockIdx.x * 256 + threadIdx.x;
    int e = gid >> 6;                 // 64 threads per edge
    if (e >= N_EDGES) return;
    int c = (gid & 63) * 4;
    int s = src[e], d = dst[e];
    float we = w[e];
    float4 v = *(const float4*)(support + (long)s * DIM + c);
    atomicAdd(&agg[(long)d * DIM + c + 0], v.x * we);
    atomicAdd(&agg[(long)d * DIM + c + 1], v.y * we);
    atomicAdd(&agg[(long)d * DIM + c + 2], v.z * we);
    atomicAdd(&agg[(long)d * DIM + c + 3], v.w * we);
}

// ---------------- bias + leaky_relu (in place on agg) ----------------------
__global__ void k_bias_relu(float* __restrict__ h, const float* __restrict__ b1) {
    int gid = blockIdx.x * 256 + threadIdx.x;
    if (gid >= N_NODES * DIM) return;
    float v = h[gid] + b1[gid & (DIM - 1)];
    h[gid] = (v > 0.0f) ? v : 0.01f * v;
}

// ---------------- GEMM2: support2[N,6] = h[N,256] @ W2[256,6] --------------
__global__ void k_gemm2(const float* __restrict__ h, const float* __restrict__ W2,
                        float* __restrict__ s2) {
    int gid = blockIdx.x * 256 + threadIdx.x;
    if (gid >= N_NODES * NOUT) return;
    int node = gid / NOUT, j = gid % NOUT;
    const float* hr = h + (long)node * DIM;
    float s = 0.0f;
#pragma unroll 8
    for (int k = 0; k < DIM; ++k) s += hr[k] * W2[k * NOUT + j];
    s2[gid] = s;
}

// ---------------- scatter layer 2 into d_out -------------------------------
__global__ void k_scatter2(const float* __restrict__ s2,
                           const int* __restrict__ src, const int* __restrict__ dst,
                           const float* __restrict__ w, float* __restrict__ out) {
    int gid = blockIdx.x * 256 + threadIdx.x;
    int e = gid >> 3, j = gid & 7;    // 8 threads per edge, 6 active
    if (e >= N_EDGES || j >= NOUT) return;
    atomicAdd(&out[(long)dst[e] * NOUT + j], s2[(long)src[e] * NOUT + j] * w[e]);
}

// ---------------- + b2 then log_softmax over 6, in place -------------------
__global__ void k_logsoftmax(float* __restrict__ out, const float* __restrict__ b2) {
    int node = blockIdx.x * 256 + threadIdx.x;
    if (node >= N_NODES) return;
    float v[NOUT];
    float m = -3.4e38f;
#pragma unroll
    for (int j = 0; j < NOUT; ++j) { v[j] = out[node * NOUT + j] + b2[j]; m = fmaxf(m, v[j]); }
    float s = 0.0f;
#pragma unroll
    for (int j = 0; j < NOUT; ++j) s += __expf(v[j] - m);
    float l = __logf(s);
#pragma unroll
    for (int j = 0; j < NOUT; ++j) out[node * NOUT + j] = v[j] - m - l;
}

extern "C" void kernel_launch(void* const* d_in, const int* in_sizes, int n_in,
                              void* d_out, int out_size, void* d_ws, size_t ws_size,
                              hipStream_t stream) {
    const float* x    = (const float*)d_in[0];
    const int*   esrc = (const int*)  d_in[1];
    const int*   edst = (const int*)  d_in[2];
    const float* ew   = (const float*)d_in[3];
    const float* W1   = (const float*)d_in[4];
    const float* b1   = (const float*)d_in[5];
    const float* W2   = (const float*)d_in[6];
    const float* b2   = (const float*)d_in[7];
    float* out = (float*)d_out;

    const size_t W1B_BYTES = (size_t)DIM * K_PAD * 2;          // 1,900,544
    const size_t BUF_BYTES = (size_t)N_NODES * DIM * 4;        // 51,200,000
    char* ws = (char*)d_ws;
    __bf16* w1bT = (__bf16*)ws;
    float*  bufA = (float*)(ws + W1B_BYTES);                   // support1 / support2
    float*  bufB = (float*)(ws + W1B_BYTES + BUF_BYTES);       // agg1 -> h

    // 1. convert W1 to transposed padded bf16 [256][3712]
    k_convert_w1<<<(DIM * K_PAD + 255) / 256, 256, 0, stream>>>(W1, w1bT);
    // 2. zero agg1
    k_zero<<<(N_NODES * DIM + 255) / 256, 256, 0, stream>>>(bufB, N_NODES * DIM);
    // 3. GEMM1 (WMMA bf16), full-N block tile -> x streamed once
    k_gemm1<<<(N_NODES + 127) / 128, 256, 0, stream>>>(x, w1bT, bufA);
    // 4. edge scatter layer 1
    k_scatter1<<<(N_EDGES * 64 + 255) / 256, 256, 0, stream>>>(bufA, esrc, edst, ew, bufB);
    // 5. bias + leaky relu -> h
    k_bias_relu<<<(N_NODES * DIM + 255) / 256, 256, 0, stream>>>(bufB, b1);
    // 6. GEMM2 (small matvec)
    k_gemm2<<<(N_NODES * NOUT + 255) / 256, 256, 0, stream>>>(bufB, W2, bufA);
    // 7. zero output (used as agg2)
    k_zero<<<(N_NODES * NOUT + 255) / 256, 256, 0, stream>>>(out, N_NODES * NOUT);
    // 8. edge scatter layer 2 into out
    k_scatter2<<<(N_EDGES * 8 + 255) / 256, 256, 0, stream>>>(bufA, esrc, edst, ew, out);
    // 9. +b2 and log_softmax, in place
    k_logsoftmax<<<(N_NODES + 255) / 256, 256, 0, stream>>>(out, b2);
}
